// LMModel_LSTM_39367670235508
// MI455X (gfx1250) — compile-verified, matrix operand order checked by reading, not run on hardware
//
#include <hip/hip_runtime.h>
#include <hip/hip_bf16.h>

// ---------------------------------------------------------------------------
// Types for CDNA5 WMMA (wave32, 16x16x32 bf16 -> f32)
// ---------------------------------------------------------------------------
typedef __bf16 bf16;
typedef __attribute__((ext_vector_type(16))) __bf16 v16bf;
typedef __attribute__((ext_vector_type(8)))  __bf16 v8bf;
typedef __attribute__((ext_vector_type(8)))  float  v8f;
typedef int v4i __attribute__((vector_size(16)));   // 16-byte chunk for async LDS copy

#define NVOC 32000
#define DIM  1024
#define HID  1024
#define TT   128
#define BB   32
#define MROWS (TT * BB)   // 4096

// ---- async global->LDS path (ASYNCcnt) availability probe ----
#if __has_builtin(__builtin_amdgcn_global_load_async_to_lds_b128)
#define HAVE_ASYNC_LDS 1
#else
#define HAVE_ASYNC_LDS 0
#endif

#if HAVE_ASYNC_LDS
// Copies 16 bytes global->LDS asynchronously; the immediate offset applies to
// both global and LDS addresses (ISA: addr = VGPR + INST_OFFSET on both sides).
// Signature (from hipcc diagnostic): arg0 = addrspace(1) v4i* (global src).
template <int OFF>
static __device__ __forceinline__ void async_ld16(const bf16* g, bf16* l) {
  __builtin_amdgcn_global_load_async_to_lds_b128(
      (__attribute__((address_space(1))) v4i*)(g),
      (__attribute__((address_space(3))) v4i*)(l),
      OFF, 0);
}
#endif

template <int N>
static __device__ __forceinline__ void wait_asynccnt() {
#if __has_builtin(__builtin_amdgcn_s_wait_asynccnt)
  __builtin_amdgcn_s_wait_asynccnt(N);
#else
  asm volatile("s_wait_asynccnt %0" ::"i"(N) : "memory");
#endif
}

// ---------------------------------------------------------------------------
// Fragment loader (A and B use the same lane pattern per CDNA5 ISA tables):
//   row/col = lane % 16
//   K elems: lanes 0-15 hold K = {0..7, 16..23}; lanes 16-31 hold {8..15, 24..31}
// base points at the (tile_row0, k0) corner of a row-major matrix, ld = K-stride.
// Works for both global and LDS-derived pointers (addrspace is inferred).
// ---------------------------------------------------------------------------
static __device__ __forceinline__ v16bf load_frag(const bf16* base, int ld, int lane) {
  const bf16* p = base + (size_t)(lane & 15) * ld + ((lane >> 4) << 3);
  union { v16bf v; v8bf h[2]; } u;
  u.h[0] = *(const v8bf*)(p);        // K = off .. off+7
  u.h[1] = *(const v8bf*)(p + 16);   // K = off+16 .. off+23
  return u.v;
}

static __device__ __forceinline__ v8f wmma_bf16(v16bf a, v16bf b, v8f c) {
  return __builtin_amdgcn_wmma_f32_16x16x32_bf16(
      /*neg_a=*/false, a, /*neg_b=*/false, b,
      /*c_mod=*/(short)0, c, /*reuse_a=*/false, /*reuse_b=*/false);
}

static __device__ __forceinline__ float sigmoidf_(float x) {
  return 1.0f / (1.0f + __expf(-x));
}

// ---------------------------------------------------------------------------
// Utility kernels
// ---------------------------------------------------------------------------
__global__ __launch_bounds__(256) void zero_u32_kernel(unsigned* p, int nwords) {
  int i = blockIdx.x * 256 + threadIdx.x;
  if (i < nwords) p[i] = 0u;
}

__global__ __launch_bounds__(256) void f32_to_bf16_kernel(const float* __restrict__ src,
                                                          bf16* __restrict__ dst, int n) {
  int i = (blockIdx.x * 256 + threadIdx.x) * 4;
  if (i + 3 < n) {
    float4 v = *(const float4*)(src + i);
    dst[i + 0] = (bf16)v.x;
    dst[i + 1] = (bf16)v.y;
    dst[i + 2] = (bf16)v.z;
    dst[i + 3] = (bf16)v.w;
  }
}

// Embedding gather: one block per (t,b) row; 256 threads x 4 elems = 1024 dims.
__global__ __launch_bounds__(256) void embed_kernel(const int* __restrict__ tokens,
                                                    const float* __restrict__ enc_w,
                                                    bf16* __restrict__ emb) {
  const int row = blockIdx.x;                 // t*B + b
  const int tok = tokens[row];
  const float* src = enc_w + (size_t)tok * DIM;
  bf16* dst = emb + (size_t)row * DIM;
  const int i = threadIdx.x * 4;
  float4 v = *(const float4*)(src + i);
  dst[i + 0] = (bf16)v.x;
  dst[i + 1] = (bf16)v.y;
  dst[i + 2] = (bf16)v.z;
  dst[i + 3] = (bf16)v.w;
}

// ---------------------------------------------------------------------------
// Layer 0 (stateless: h_prev == c_prev == 0, f-gate is dead):
//   h0 = sigmoid(x@uo^T + wo_b+uo_b) * tanh( sigmoid(x@ui^T + wi_b+ui_b)
//                                            * tanh(x@uc^T + wc_b+uc_b) )
// grid = (MROWS/32, HID/32), block = 128 (4 waves); each wave owns a 16x16
// output patch and accumulates the 3 live gates over K=1024.
// ---------------------------------------------------------------------------
__global__ __launch_bounds__(128) void layer0_kernel(
    const bf16* __restrict__ emb,
    const bf16* __restrict__ ui, const bf16* __restrict__ uc, const bf16* __restrict__ uo,
    const float* __restrict__ wi_b, const float* __restrict__ ui_b,
    const float* __restrict__ wc_b, const float* __restrict__ uc_b,
    const float* __restrict__ wo_b, const float* __restrict__ uo_b,
    bf16* __restrict__ h0) {
  const int lane = threadIdx.x & 31;
  const int wv   = threadIdx.x >> 5;
  const int m0 = blockIdx.x * 32 + ((wv >> 1) << 4);
  const int n0 = blockIdx.y * 32 + ((wv & 1) << 4);

  v8f acc_i = {}, acc_c = {}, acc_o = {};
  for (int k0 = 0; k0 < DIM; k0 += 32) {
    v16bf a  = load_frag(emb + (size_t)m0 * DIM + k0, DIM, lane);
    v16bf bi = load_frag(ui  + (size_t)n0 * DIM + k0, DIM, lane);
    v16bf bc = load_frag(uc  + (size_t)n0 * DIM + k0, DIM, lane);
    v16bf bo = load_frag(uo  + (size_t)n0 * DIM + k0, DIM, lane);
    acc_i = wmma_bf16(a, bi, acc_i);
    acc_c = wmma_bf16(a, bc, acc_c);
    acc_o = wmma_bf16(a, bo, acc_o);
  }

  const int n     = n0 + (lane & 15);
  const int mbase = m0 + ((lane >> 4) << 3);
  const float bi_ = wi_b[n] + ui_b[n];
  const float bc_ = wc_b[n] + uc_b[n];
  const float bo_ = wo_b[n] + uo_b[n];
#pragma unroll
  for (int r = 0; r < 8; ++r) {
    const float ig = sigmoidf_(acc_i[r] + bi_);
    const float cg = tanhf(acc_c[r] + bc_);
    const float og = sigmoidf_(acc_o[r] + bo_);
    const float h  = og * tanhf(ig * cg);
    h0[(size_t)(mbase + r) * HID + n] = (bf16)h;
  }
}

// ---------------------------------------------------------------------------
// Layer 1 recurrent step. grid = HID/32 blocks, block = 128 (4 waves).
// Each wave owns a 16(batch) x 16(hidden) patch and accumulates all four
// gates over K = 1024 (h_prev @ W^T) + 1024 (x @ U^T), then updates c/h.
// ---------------------------------------------------------------------------
__global__ __launch_bounds__(128) void rnn_step_kernel(
    const bf16* __restrict__ h_prev,   // [B, HID] bf16
    const bf16* __restrict__ x,        // [B, HID] bf16 (layer-0 output at t)
    float*      __restrict__ c_state,  // [B, HID] f32, in/out
    bf16*       __restrict__ h_out,    // [B, HID] bf16 (ys slice at t)
    const bf16* __restrict__ wf, const bf16* __restrict__ wi,
    const bf16* __restrict__ wc, const bf16* __restrict__ wo,
    const bf16* __restrict__ uf, const bf16* __restrict__ ui,
    const bf16* __restrict__ uc, const bf16* __restrict__ uo,
    const float* __restrict__ wf_b, const float* __restrict__ uf_b,
    const float* __restrict__ wi_b, const float* __restrict__ ui_b,
    const float* __restrict__ wc_b, const float* __restrict__ uc_b,
    const float* __restrict__ wo_b, const float* __restrict__ uo_b) {
  const int lane = threadIdx.x & 31;
  const int wv   = threadIdx.x >> 5;
  const int m0 = ((wv >> 1) << 4);                 // batch tile: 0 or 16
  const int n0 = blockIdx.x * 32 + ((wv & 1) << 4);

  v8f af = {}, ai = {}, ac = {}, ao = {};
  // h_prev @ {wf,wi,wc,wo}^T
  for (int k0 = 0; k0 < HID; k0 += 32) {
    v16bf a = load_frag(h_prev + (size_t)m0 * HID + k0, HID, lane);
    af = wmma_bf16(a, load_frag(wf + (size_t)n0 * HID + k0, HID, lane), af);
    ai = wmma_bf16(a, load_frag(wi + (size_t)n0 * HID + k0, HID, lane), ai);
    ac = wmma_bf16(a, load_frag(wc + (size_t)n0 * HID + k0, HID, lane), ac);
    ao = wmma_bf16(a, load_frag(wo + (size_t)n0 * HID + k0, HID, lane), ao);
  }
  // x @ {uf,ui,uc,uo}^T
  for (int k0 = 0; k0 < DIM; k0 += 32) {
    v16bf a = load_frag(x + (size_t)m0 * DIM + k0, DIM, lane);
    af = wmma_bf16(a, load_frag(uf + (size_t)n0 * DIM + k0, DIM, lane), af);
    ai = wmma_bf16(a, load_frag(ui + (size_t)n0 * DIM + k0, DIM, lane), ai);
    ac = wmma_bf16(a, load_frag(uc + (size_t)n0 * DIM + k0, DIM, lane), ac);
    ao = wmma_bf16(a, load_frag(uo + (size_t)n0 * DIM + k0, DIM, lane), ao);
  }

  const int n     = n0 + (lane & 15);
  const int mbase = m0 + ((lane >> 4) << 3);
  const float bf_ = wf_b[n] + uf_b[n];
  const float bi_ = wi_b[n] + ui_b[n];
  const float bc_ = wc_b[n] + uc_b[n];
  const float bo_ = wo_b[n] + uo_b[n];
#pragma unroll
  for (int r = 0; r < 8; ++r) {
    const size_t idx = (size_t)(mbase + r) * HID + n;
    const float fg = sigmoidf_(af[r] + bf_);
    const float ig = sigmoidf_(ai[r] + bi_);
    const float cg = tanhf(ac[r] + bc_);
    const float og = sigmoidf_(ao[r] + bo_);
    const float cn = fg * c_state[idx] + ig * cg;
    c_state[idx] = cn;
    h_out[idx] = (bf16)(og * tanhf(cn));
  }
}

// ---------------------------------------------------------------------------
// Decoder GEMM: out[M=4096, N=32000] = ys[4096,1024] @ dec_w[32000,1024]^T + b
//
// grid = (M/64, N/256), block = 256 (8 waves as 2(M) x 4(N)); each wave
// computes a 32x64 tile (2x4 WMMA tiles, 8 f32 accumulators).
// A (ys) is double-buffered in LDS via ASYNC global->LDS copies (ASYNCcnt):
// the next K-block's copies are issued before computing the current block,
// then s_wait_asynccnt leaves only the newest 2 in flight (async loads
// complete in order per wave) + barrier. B (dec_w, bf16, 65 MB) streams from
// L2 (fits in MI455X's 192 MB). Output (524 MB, write-once) uses non-temporal
// stores so it does not evict dec_w from L2.
// ---------------------------------------------------------------------------
#define DEC_BK   64
#define DEC_LDA  (DEC_BK + 8)   // padded row stride (elems) to dodge LDS bank conflicts

__global__ __launch_bounds__(256) void decoder_kernel(
    const bf16* __restrict__ ys, const bf16* __restrict__ decw,
    const float* __restrict__ dec_b, float* __restrict__ out) {
  __shared__ __attribute__((aligned(16))) bf16 As[2][64][DEC_LDA];

  const int lane = threadIdx.x & 31;
  const int wv   = threadIdx.x >> 5;
  const int mw   = wv >> 2;                         // 0..1  (M wave row)
  const int nw   = wv & 3;                          // 0..3  (N wave col)
  const int m0   = blockIdx.x * 64;
  const int n0   = blockIdx.y * 256 + nw * 64;
  const int wm   = mw * 32;                         // wave M offset in block

  // staging assignment: 256 threads cover 64 rows x 64 cols, 16 elems each
  const int srow = threadIdx.x >> 2;
  const int scol = (threadIdx.x & 3) * 16;

  // stage ys[m0:m0+64, kb:kb+BK] -> As[buf]  (two 16-B chunks per thread)
  auto stage = [&](int buf, int kb) {
    const bf16* src = ys + (size_t)(m0 + srow) * HID + kb + scol;
    bf16* dst = &As[buf][srow][scol];
#if HAVE_ASYNC_LDS
    async_ld16<0>(src, dst);
    async_ld16<16>(src, dst);
#else
    *(v8bf*)(dst)     = *(const v8bf*)(src);
    *(v8bf*)(dst + 8) = *(const v8bf*)(src + 8);
#endif
  };

  v8f acc[2][4];
#pragma unroll
  for (int mi = 0; mi < 2; ++mi)
#pragma unroll
    for (int ni = 0; ni < 4; ++ni) acc[mi][ni] = (v8f){};

  stage(0, 0);                                      // prologue: first block

  int buf = 0;
  for (int kb = 0; kb < HID; kb += DEC_BK, buf ^= 1) {
    const bool more = (kb + DEC_BK) < HID;
    if (more) stage(buf ^ 1, kb + DEC_BK);          // async into other buffer

#if HAVE_ASYNC_LDS
    if (more) wait_asynccnt<2>();                   // current block landed
    else      wait_asynccnt<0>();
#endif
    __syncthreads();                                // LDS visible WG-wide

    // prefetch next K-block of B one block ahead
    if (more) {
#pragma unroll
      for (int ni = 0; ni < 4; ++ni) {
        const bf16* pf = decw + (size_t)(n0 + ni * 16 + (lane & 15)) * HID +
                         (kb + DEC_BK) + ((lane >> 4) << 3);
        __builtin_prefetch(pf, 0, 1);
      }
    }

#pragma unroll
    for (int kk = 0; kk < DEC_BK; kk += 32) {
      v16bf a0 = load_frag(&As[buf][wm][kk],      DEC_LDA, lane);
      v16bf a1 = load_frag(&As[buf][wm + 16][kk], DEC_LDA, lane);
      v16bf b[4];
#pragma unroll
      for (int ni = 0; ni < 4; ++ni)
        b[ni] = load_frag(decw + (size_t)(n0 + ni * 16) * HID + kb + kk, HID, lane);
#pragma unroll
      for (int ni = 0; ni < 4; ++ni) {
        acc[0][ni] = wmma_bf16(a0, b[ni], acc[0][ni]);
        acc[1][ni] = wmma_bf16(a1, b[ni], acc[1][ni]);
      }
    }
    __syncthreads();                                // done reading As[buf]
  }

  // ---- epilogue: bias + non-temporal f32 stores ----
  const int mb = wm + ((lane >> 4) << 3);
  float bias[4];
#pragma unroll
  for (int ni = 0; ni < 4; ++ni) bias[ni] = dec_b[n0 + ni * 16 + (lane & 15)];

#pragma unroll
  for (int mi = 0; mi < 2; ++mi) {
#pragma unroll
    for (int r = 0; r < 8; ++r) {
      const size_t row = (size_t)(m0 + mb + mi * 16 + r) * NVOC;
#pragma unroll
      for (int ni = 0; ni < 4; ++ni) {
        const int col = n0 + ni * 16 + (lane & 15);
        __builtin_nontemporal_store(acc[mi][ni][r] + bias[ni], &out[row + col]);
      }
    }
  }
}

// ---------------------------------------------------------------------------
// Host orchestration
// ---------------------------------------------------------------------------
extern "C" void kernel_launch(void* const* d_in, const int* in_sizes, int n_in,
                              void* d_out, int out_size, void* d_ws, size_t ws_size,
                              hipStream_t stream) {
  const int*   tokens = (const int*)d_in[0];
  const float* enc_w  = (const float*)d_in[1];
  const float* wf_w   = (const float*)d_in[2];
  const float* wf_b   = (const float*)d_in[3];
  const float* uf_w   = (const float*)d_in[4];
  const float* uf_b   = (const float*)d_in[5];
  const float* wi_w   = (const float*)d_in[6];
  const float* wi_b   = (const float*)d_in[7];
  const float* ui_w   = (const float*)d_in[8];
  const float* ui_b   = (const float*)d_in[9];
  const float* wc_w   = (const float*)d_in[10];
  const float* wc_b   = (const float*)d_in[11];
  const float* uc_w   = (const float*)d_in[12];
  const float* uc_b   = (const float*)d_in[13];
  const float* wo_w   = (const float*)d_in[14];
  const float* wo_b   = (const float*)d_in[15];
  const float* uo_w   = (const float*)d_in[16];
  const float* uo_b   = (const float*)d_in[17];
  const float* dec_w  = (const float*)d_in[18];
  const float* dec_b  = (const float*)d_in[19];
  float* out = (float*)d_out;

  // -------- workspace carve-out (aligned to 256 B) --------
  size_t off = 0;
  auto carve = [&](size_t bytes) -> void* {
    void* r = (char*)d_ws + off;
    off += (bytes + 255) & ~(size_t)255;
    return r;
  };
  bf16*  emb   = (bf16*)carve((size_t)MROWS * DIM * sizeof(bf16));
  bf16*  h0    = (bf16*)carve((size_t)MROWS * HID * sizeof(bf16));
  bf16*  ys    = (bf16*)carve((size_t)MROWS * HID * sizeof(bf16));
  float* cst   = (float*)carve((size_t)BB * HID * sizeof(float));
  bf16*  hzero = (bf16*)carve((size_t)BB * HID * sizeof(bf16));
  bf16*  bwf = (bf16*)carve((size_t)HID * HID * sizeof(bf16));
  bf16*  bwi = (bf16*)carve((size_t)HID * HID * sizeof(bf16));
  bf16*  bwc = (bf16*)carve((size_t)HID * HID * sizeof(bf16));
  bf16*  bwo = (bf16*)carve((size_t)HID * HID * sizeof(bf16));
  bf16*  buf = (bf16*)carve((size_t)HID * DIM * sizeof(bf16));
  bf16*  bui = (bf16*)carve((size_t)HID * DIM * sizeof(bf16));
  bf16*  buc = (bf16*)carve((size_t)HID * DIM * sizeof(bf16));
  bf16*  buo = (bf16*)carve((size_t)HID * DIM * sizeof(bf16));
  bf16*  bdec = (bf16*)carve((size_t)NVOC * HID * sizeof(bf16));

  // -------- weight conversions f32 -> bf16 --------
  const int wElems = HID * DIM;                       // 1M elems
  const int wBlocks = wElems / (256 * 4);             // 1024 blocks
  f32_to_bf16_kernel<<<wBlocks, 256, 0, stream>>>(wf_w, bwf, wElems);
  f32_to_bf16_kernel<<<wBlocks, 256, 0, stream>>>(wi_w, bwi, wElems);
  f32_to_bf16_kernel<<<wBlocks, 256, 0, stream>>>(wc_w, bwc, wElems);
  f32_to_bf16_kernel<<<wBlocks, 256, 0, stream>>>(wo_w, bwo, wElems);
  f32_to_bf16_kernel<<<wBlocks, 256, 0, stream>>>(uf_w, buf, wElems);
  f32_to_bf16_kernel<<<wBlocks, 256, 0, stream>>>(ui_w, bui, wElems);
  f32_to_bf16_kernel<<<wBlocks, 256, 0, stream>>>(uc_w, buc, wElems);
  f32_to_bf16_kernel<<<wBlocks, 256, 0, stream>>>(uo_w, buo, wElems);
  const int dElems = NVOC * HID;                      // 32.77M elems
  f32_to_bf16_kernel<<<dElems / (256 * 4), 256, 0, stream>>>(dec_w, bdec, dElems);

  // -------- zero-init recurrent state --------
  {
    const int cw = BB * HID;                          // f32 words
    zero_u32_kernel<<<(cw + 255) / 256, 256, 0, stream>>>((unsigned*)cst, cw);
    const int hw = BB * HID / 2;                      // bf16 pairs as u32 words
    zero_u32_kernel<<<(hw + 255) / 256, 256, 0, stream>>>((unsigned*)hzero, hw);
  }

  // -------- embedding gather --------
  embed_kernel<<<MROWS, 256, 0, stream>>>(tokens, enc_w, emb);

  // -------- layer 0 (stateless, fully batched over T*B) --------
  {
    dim3 grid(MROWS / 32, HID / 32);
    layer0_kernel<<<grid, 128, 0, stream>>>(emb, bui, buc, buo,
                                            wi_b, ui_b, wc_b, uc_b, wo_b, uo_b, h0);
  }

  // -------- layer 1 (sequential recurrence over T) --------
  for (int t = 0; t < TT; ++t) {
    const bf16* hp = (t == 0) ? hzero : (ys + (size_t)(t - 1) * BB * HID);
    const bf16* xt = h0 + (size_t)t * BB * HID;
    bf16*       ho = ys + (size_t)t * BB * HID;
    rnn_step_kernel<<<HID / 32, 128, 0, stream>>>(
        hp, xt, cst, ho,
        bwf, bwi, bwc, bwo, buf, bui, buc, buo,
        wf_b, uf_b, wi_b, ui_b, wc_b, uc_b, wo_b, uo_b);
  }

  // -------- decoder --------
  {
    dim3 grid(MROWS / 64, NVOC / 256);
    decoder_kernel<<<grid, 256, 0, stream>>>(ys, bdec, dec_b, out);
  }
}
